// GraphTransformerNetwork_50921132261401
// MI455X (gfx1250) — matrix operand order repaired and checked by reading, not compiled, not run on hardware
//
#include <hip/hip_runtime.h>
#include <hip/hip_bf16.h>

// ---------------------------------------------------------------------------
// Problem constants (match reference)
// ---------------------------------------------------------------------------
#define NN   50000      // nodes
#define EE   800000     // edges
#define DD   64         // input dim
#define HID  64         // hidden per head
#define NH   4          // heads
#define GG   128        // graphs
#define AA   32         // actions
#define NC   832        // packed projection width: q(256)+k(256)+v(256)+skip(64)

typedef __attribute__((ext_vector_type(16))) __bf16 v16bf;
typedef __attribute__((ext_vector_type(8)))  float  v8f;

union V16 { v16bf v; uint4 q[2]; };

// float -> bf16 bits, round-to-nearest-even
static __device__ __forceinline__ unsigned short f2bf_bits(float f) {
    unsigned u = __float_as_uint(f);
    unsigned r = (u + 0x7FFFu + ((u >> 16) & 1u)) >> 16;
    return (unsigned short)r;
}

// monotone float<->uint encoding for atomicMax-based segment max
static __device__ __forceinline__ unsigned fenc(float f) {
    unsigned u = __float_as_uint(f);
    return (u & 0x80000000u) ? ~u : (u | 0x80000000u);
}
static __device__ __forceinline__ float fdec(unsigned e) {
    unsigned u = (e & 0x80000000u) ? (e ^ 0x80000000u) : ~e;
    return __uint_as_float(u);
}
#define ENC_NEG_INF 0x007FFFFFu   // fenc(-inf)

// ---------------------------------------------------------------------------
// Utility fills
// ---------------------------------------------------------------------------
__global__ void fill_f32(float* p, float v, long n) {
    long i = (long)blockIdx.x * blockDim.x + threadIdx.x;
    if (i < n) p[i] = v;
}
__global__ void fill_u32(unsigned* p, unsigned v, long n) {
    long i = (long)blockIdx.x * blockDim.x + threadIdx.x;
    if (i < n) p[i] = v;
}

// convert fp32 [n] -> bf16 bits [n]
__global__ void cvt_bf16(const float* __restrict__ x, unsigned short* __restrict__ y, long n) {
    long i = (long)blockIdx.x * blockDim.x + threadIdx.x;
    if (i < n) y[i] = f2bf_bits(x[i]);
}

// pack Wq|Wk|Wv|Ws -> bf16 [NC][64], biases -> fp32 [NC]
__global__ void pack_w(const float* __restrict__ Wq, const float* __restrict__ bq,
                       const float* __restrict__ Wk, const float* __restrict__ bk,
                       const float* __restrict__ Wv, const float* __restrict__ bv,
                       const float* __restrict__ Ws, const float* __restrict__ bs,
                       unsigned short* __restrict__ Wp, float* __restrict__ bp) {
    int i = blockIdx.x * blockDim.x + threadIdx.x;
    if (i < NC * 64) {
        int row = i >> 6, col = i & 63;
        float v;
        if      (row < 256) v = Wq[row * 64 + col];
        else if (row < 512) v = Wk[(row - 256) * 64 + col];
        else if (row < 768) v = Wv[(row - 512) * 64 + col];
        else                v = Ws[(row - 768) * 64 + col];
        Wp[i] = f2bf_bits(v);
    }
    if (i < NC) {
        float b;
        if      (i < 256) b = bq[i];
        else if (i < 512) b = bk[i - 256];
        else if (i < 768) b = bv[i - 512];
        else              b = bs[i - 768];
        bp[i] = b;
    }
}

// ---------------------------------------------------------------------------
// Projection GEMM: Y[M,NC] = Xbf[M,64] * Wp[NC,64]^T + bp  (bf16 WMMA, f32 acc)
// One 16x16 output tile per wave32; K=64 -> two v_wmma_f32_16x16x32_bf16.
// ---------------------------------------------------------------------------
__global__ void wmma_proj_gemm(const unsigned short* __restrict__ Xbf,
                               const unsigned short* __restrict__ Wp,
                               const float* __restrict__ bp,
                               float* __restrict__ Y, int M) {
    const int ntN   = NC >> 4;                       // 52 column tiles
    const int total = (M >> 4) * ntN;
    int wave = (blockIdx.x * blockDim.x + threadIdx.x) >> 5;
    if (wave >= total) return;
    int lane = threadIdx.x & 31;
    int mt = wave / ntN, nt = wave - mt * ntN;
    int half = lane >> 4, r = lane & 15;

    v8f c = {};
    V16 a, b;
#pragma unroll
    for (int kb = 0; kb < 2; ++kb) {
        // A fragment: lane = row (mt*16+r); elems 0-7 -> K = kb*32+8*half+[0,8)
        //                                    elems 8-15 -> K = kb*32+16+8*half+[0,8)
        const uint4* pa = (const uint4*)(Xbf + (size_t)(mt * 16 + r) * 64 + kb * 32 + 8 * half);
        a.q[0] = pa[0];
        a.q[1] = pa[2];                              // +16 elements
        // B fragment: lane = out-col (nt*16+r); elems 0-7 -> K = kb*32+8*half+[0,8), elems 8-15 -> K = kb*32+16+8*half+[0,8)
        const uint4* pb = (const uint4*)(Wp + (size_t)(nt * 16 + r) * 64 + kb * 32 + 8 * half);
        b.q[0] = pb[0];
        b.q[1] = pb[2];
        c = __builtin_amdgcn_wmma_f32_16x16x32_bf16(false, a.v, false, b.v,
                                                    (short)0, c, false, false);
    }
    // C/D layout: VGPR i -> row = mt*16 + i + 8*half, col = nt*16 + (lane&15)
    int col = nt * 16 + r;
    float bias = bp[col];
    float* yb = Y + (size_t)(mt * 16 + 8 * half) * NC + col;
#pragma unroll
    for (int i = 0; i < 8; ++i) yb[(size_t)i * NC] = c[i] + bias;
}

// ---------------------------------------------------------------------------
// Attention: one wave per edge. alpha[e][h] = <q[dst],k[src]>_h / sqrt(HID)
// ---------------------------------------------------------------------------
__global__ void attn_logits(const float* __restrict__ proj,
                            const int* __restrict__ src, const int* __restrict__ dst,
                            float* __restrict__ alpha, unsigned* __restrict__ mEnc) {
    int e = (blockIdx.x * blockDim.x + threadIdx.x) >> 5;
    if (e >= EE) return;
    int lane = threadIdx.x & 31;
    int h = lane >> 3, sub = lane & 7;               // 8 lanes per head, 8 chans/lane
    int s = src[e], d = dst[e];
    const float* q = proj + (size_t)d * NC + 0   + h * 64 + sub * 8;
    const float* k = proj + (size_t)s * NC + 256 + h * 64 + sub * 8;
    float acc = 0.f;
#pragma unroll
    for (int i = 0; i < 8; ++i) acc += q[i] * k[i];
    acc += __shfl_xor(acc, 1, 32);
    acc += __shfl_xor(acc, 2, 32);
    acc += __shfl_xor(acc, 4, 32);
    if (sub == 0) {
        float a = acc * 0.125f;                      // 1/sqrt(64)
        alpha[(size_t)e * NH + h] = a;
        atomicMax(&mEnc[(size_t)d * NH + h], fenc(a));
    }
}

// ea = exp(alpha - m[dst]); z[dst] += ea   (thread per (edge,head); in-place)
__global__ void attn_exp(const int* __restrict__ dst, float* __restrict__ alpha,
                         const unsigned* __restrict__ mEnc, float* __restrict__ z) {
    long idx = (long)blockIdx.x * blockDim.x + threadIdx.x;
    if (idx >= (long)EE * NH) return;
    int e = (int)(idx >> 2), h = (int)(idx & 3);
    int d = dst[e];
    float m = fdec(mEnc[(size_t)d * NH + h]);
    float v = __expf(alpha[idx] - m);
    alpha[idx] = v;                                  // now holds ea
    atomicAdd(&z[(size_t)d * NH + h], v);
}

// agg[dst][h*64+c] += (ea/z) * v[src][h*64+c]   (wave per edge, 8 chans/lane)
__global__ void attn_agg(const float* __restrict__ proj, const float* __restrict__ ea,
                         const float* __restrict__ z,
                         const int* __restrict__ src, const int* __restrict__ dst,
                         float* __restrict__ agg) {
    int e = (blockIdx.x * blockDim.x + threadIdx.x) >> 5;
    if (e >= EE) return;
    int lane = threadIdx.x & 31;
    int h = lane >> 3;
    int s = src[e], d = dst[e];
    float a = ea[(size_t)e * NH + h] / (z[(size_t)d * NH + h] + 1e-16f);
    const float* vrow = proj + (size_t)s * NC + 512 + lane * 8;
    float* orow = agg + (size_t)d * (NH * HID) + lane * 8;
#pragma unroll
    for (int i = 0; i < 8; ++i) atomicAdd(&orow[i], a * vrow[i]);
}

// hout = relu(head_mean(agg) + skip)   (thread per (n,c))
__global__ void combine(const float* __restrict__ agg, const float* __restrict__ proj,
                        float* __restrict__ hout) {
    long idx = (long)blockIdx.x * blockDim.x + threadIdx.x;
    if (idx >= (long)NN * HID) return;
    long n = idx >> 6; int c = (int)(idx & 63);
    const float* ar = agg + n * (NH * HID);
    float m = 0.25f * (ar[c] + ar[64 + c] + ar[128 + c] + ar[192 + c]);
    float v = m + proj[(size_t)n * NC + 768 + c];
    hout[idx] = v > 0.f ? v : 0.f;
}

// global mean pool accumulation
__global__ void pool_sum(const float* __restrict__ h, const int* __restrict__ batch,
                         float* __restrict__ gsum, float* __restrict__ gcnt) {
    long idx = (long)blockIdx.x * blockDim.x + threadIdx.x;
    if (idx >= (long)NN * HID) return;
    long n = idx >> 6; int c = (int)(idx & 63);
    int b = batch[n];
    atomicAdd(&gsum[b * HID + c], h[idx]);
    if (c == 0) atomicAdd(&gcnt[b], 1.0f);
}

// dueling heads: one block (64 threads) per graph
__global__ void heads(const float* __restrict__ gsum, const float* __restrict__ gcnt,
                      const float* __restrict__ W1v, const float* __restrict__ b1v,
                      const float* __restrict__ W2v, const float* __restrict__ b2v,
                      const float* __restrict__ W1a, const float* __restrict__ b1a,
                      const float* __restrict__ W2a, const float* __restrict__ b2a,
                      float* __restrict__ out) {
    __shared__ float g[64], hv[64], ha[64], adv[32], red[64];
    int gi = blockIdx.x, t = threadIdx.x;
    float cnt = gcnt[gi]; if (cnt < 1.f) cnt = 1.f;
    g[t] = gsum[gi * 64 + t] / cnt;
    __syncthreads();
    float av = b1v[t], aa = b1a[t];
#pragma unroll 8
    for (int i = 0; i < 64; ++i) { av += g[i] * W1v[t * 64 + i]; aa += g[i] * W1a[t * 64 + i]; }
    hv[t] = av > 0.f ? av : 0.f;
    ha[t] = aa > 0.f ? aa : 0.f;
    __syncthreads();
    red[t] = hv[t] * W2v[t];
    __syncthreads();
    for (int s = 32; s > 0; s >>= 1) { if (t < s) red[t] += red[t + s]; __syncthreads(); }
    float val = red[0] + b2v[0];
    if (t < AA) {
        float a = b2a[t];
#pragma unroll 8
        for (int i = 0; i < 64; ++i) a += ha[i] * W2a[t * 64 + i];
        adv[t] = a;
    }
    __syncthreads();
    if (t == 0) { float s = 0.f; for (int i = 0; i < AA; ++i) s += adv[i]; red[0] = s * (1.f / AA); }
    __syncthreads();
    if (t < AA) out[gi * AA + t] = val + adv[t] - red[0];
}

// ---------------------------------------------------------------------------
// Host side
// ---------------------------------------------------------------------------
static inline unsigned gdim(long n, int b) { return (unsigned)((n + b - 1) / b); }

extern "C" void kernel_launch(void* const* d_in, const int* in_sizes, int n_in,
                              void* d_out, int out_size, void* d_ws, size_t ws_size,
                              hipStream_t stream) {
    const float* x     = (const float*)d_in[0];
    const int*   eidx  = (const int*)d_in[1];
    const int*   batch = (const int*)d_in[2];
    const int*   src   = eidx;
    const int*   dst   = eidx + EE;
    const float* Wq1 = (const float*)d_in[3],  *bq1 = (const float*)d_in[4];
    const float* Wk1 = (const float*)d_in[5],  *bk1 = (const float*)d_in[6];
    const float* Wv1 = (const float*)d_in[7],  *bv1 = (const float*)d_in[8];
    const float* Ws1 = (const float*)d_in[9],  *bs1 = (const float*)d_in[10];
    const float* Wq2 = (const float*)d_in[11], *bq2 = (const float*)d_in[12];
    const float* Wk2 = (const float*)d_in[13], *bk2 = (const float*)d_in[14];
    const float* Wv2 = (const float*)d_in[15], *bv2 = (const float*)d_in[16];
    const float* Ws2 = (const float*)d_in[17], *bs2 = (const float*)d_in[18];
    const float* W1v = (const float*)d_in[19], *b1v = (const float*)d_in[20];
    const float* W2v = (const float*)d_in[21], *b2v = (const float*)d_in[22];
    const float* W1a = (const float*)d_in[23], *b1a = (const float*)d_in[24];
    const float* W2a = (const float*)d_in[25], *b2a = (const float*)d_in[26];
    float* out = (float*)d_out;

    // workspace layout (bytes)
    char* ws = (char*)d_ws;
    size_t off = 0;
    auto take = [&](size_t bytes) { char* p = ws + off; off += (bytes + 255) & ~255ull; return p; };
    unsigned short* WP1 = (unsigned short*)take((size_t)NC * 64 * 2);
    float*          BP1 = (float*)take((size_t)NC * 4);
    unsigned short* WP2 = (unsigned short*)take((size_t)NC * 64 * 2);
    float*          BP2 = (float*)take((size_t)NC * 4);
    unsigned short* XBF = (unsigned short*)take((size_t)NN * 64 * 2);
    float*          PROJ= (float*)take((size_t)NN * NC * 4);
    float*          AGG = (float*)take((size_t)NN * NH * HID * 4);
    unsigned*       MENC= (unsigned*)take((size_t)NN * NH * 4);
    float*          Z   = (float*)take((size_t)NN * NH * 4);
    float*          ALPH= (float*)take((size_t)EE * NH * 4);
    float*          H1  = (float*)take((size_t)NN * HID * 4);
    float*          H2  = (float*)take((size_t)NN * HID * 4);
    float*          GSUM= (float*)take((size_t)GG * HID * 4);
    float*          GCNT= (float*)take((size_t)GG * 4);

    const long nNC   = (long)NN * HID;              // 3.2M elementwise size
    const long gemmT = (long)(NN / 16) * (NC / 16) * 32;  // GEMM threads
    const long edgeT = (long)EE * 32;               // wave-per-edge threads

    auto run_layer = [&](const float* xin,
                         const float* Wq, const float* bq, const float* Wk, const float* bk,
                         const float* Wv, const float* bv, const float* Wsk, const float* bsk,
                         unsigned short* WP, float* BP, float* hout) {
        pack_w<<<gdim(NC * 64, 256), 256, 0, stream>>>(Wq, bq, Wk, bk, Wv, bv, Wsk, bsk, WP, BP);
        cvt_bf16<<<gdim(nNC, 256), 256, 0, stream>>>(xin, XBF, nNC);
        fill_u32<<<gdim((long)NN * NH, 256), 256, 0, stream>>>(MENC, ENC_NEG_INF, (long)NN * NH);
        fill_f32<<<gdim((long)NN * NH, 256), 256, 0, stream>>>(Z, 0.f, (long)NN * NH);
        fill_f32<<<gdim((long)NN * NH * HID, 256), 256, 0, stream>>>(AGG, 0.f, (long)NN * NH * HID);
        wmma_proj_gemm<<<gdim(gemmT, 256), 256, 0, stream>>>(XBF, WP, BP, PROJ, NN);
        attn_logits<<<gdim(edgeT, 256), 256, 0, stream>>>(PROJ, src, dst, ALPH, MENC);
        attn_exp<<<gdim((long)EE * NH, 256), 256, 0, stream>>>(dst, ALPH, MENC, Z);
        attn_agg<<<gdim(edgeT, 256), 256, 0, stream>>>(PROJ, ALPH, Z, src, dst, AGG);
        combine<<<gdim(nNC, 256), 256, 0, stream>>>(AGG, PROJ, hout);
    };

    run_layer(x,  Wq1, bq1, Wk1, bk1, Wv1, bv1, Ws1, bs1, WP1, BP1, H1);
    run_layer(H1, Wq2, bq2, Wk2, bk2, Wv2, bv2, Ws2, bs2, WP2, BP2, H2);

    fill_f32<<<gdim((long)GG * HID, 256), 256, 0, stream>>>(GSUM, 0.f, (long)GG * HID);
    fill_f32<<<gdim(GG, 256), 256, 0, stream>>>(GCNT, 0.f, GG);
    pool_sum<<<gdim(nNC, 256), 256, 0, stream>>>(H2, batch, GSUM, GCNT);
    heads<<<GG, 64, 0, stream>>>(GSUM, GCNT, W1v, b1v, W2v, b2v, W1a, b1a, W2a, b2a, out);
}
